// Discriminator_4L_51316269253413
// MI455X (gfx1250) — matrix operand order repaired and checked by reading, not compiled, not run on hardware
//
#include <hip/hip_runtime.h>
#include <hip/hip_bf16.h>

// CDNA5 / gfx1250 implementation.
// label stats -> conv1 (VALU f32->f16) -> conv2/3/4 (implicit-GEMM WMMA, LDS-staged
// im2col A-tiles + padded N-major f16 weights) -> feature@T (WMMA, TDM-staged B panel)
// -> pairwise L1+exp (VALU, b128 loads) -> sigmoid dot.

typedef __attribute__((ext_vector_type(16))) _Float16 v16h;
typedef __attribute__((ext_vector_type(8)))  _Float16 v8h;
typedef __attribute__((ext_vector_type(8)))  float    v8f;
typedef __attribute__((ext_vector_type(4)))  unsigned int v4u;
typedef __attribute__((ext_vector_type(8)))  int      v8i;
typedef __attribute__((ext_vector_type(4)))  int      v4i;

#define BATCH 512
#define FEAT  325      // 256 conv features + 5 label (+ 64 out_T appended)
#define FEATK 261
#define FEATPAD 288    // 261 padded to multiple of 32
#define MDIM  1024

#if defined(__has_builtin)
#if __has_builtin(__builtin_amdgcn_tensor_load_to_lds) && __has_builtin(__builtin_amdgcn_s_wait_tensorcnt)
#define USE_TDM 1
#endif
#endif
#ifndef USE_TDM
#define USE_TDM 0
#endif

// Build a v16h WMMA fragment from two contiguous 8-half (16B) runs.
__device__ __forceinline__ v16h make_frag(const _Float16* p0, const _Float16* p1)
{
    v8h lo = *(const v8h*)p0;
    v8h hi = *(const v8h*)p1;
    return __builtin_shufflevector(lo, hi, 0,1,2,3,4,5,6,7,8,9,10,11,12,13,14,15);
}

// ---------------------------------------------------------------- prep: padded f16 weights
// w{2,3,4}p stored (COUT x KPAD) row-major, zero padded; ThT stored N-major (1024 x 288).
__global__ __launch_bounds__(256)
void prep_kernel(const float* __restrict__ w2, const float* __restrict__ w3,
                 const float* __restrict__ w4, const float* __restrict__ T,
                 _Float16* __restrict__ w2p, _Float16* __restrict__ w3p,
                 _Float16* __restrict__ w4p, _Float16* __restrict__ ThT)
{
    const int i = blockIdx.x * blockDim.x + threadIdx.x;
    if (i < 32 * 416) {
        int n = i / 416, k = i % 416;
        w2p[i] = (k < 400) ? (_Float16)w2[n * 400 + k] : (_Float16)0.f;
    }
    if (i < 48 * 800) {
        w3p[i] = (_Float16)w3[i];                       // 800 is already a multiple of 32
    }
    if (i < 64 * 448) {
        int n = i / 448, k = i % 448;
        w4p[i] = (k < 432) ? (_Float16)w4[n * 432 + k] : (_Float16)0.f;
    }
    if (i < MDIM * FEATPAD) {
        int n = i / FEATPAD, k = i % FEATPAD;
        ThT[i] = (k < FEATK) ? (_Float16)T[(size_t)k * MDIM + n] : (_Float16)0.f;
    }
}

// ---------------------------------------------------------------- conv1: direct, f32 in, f16 out
__global__ __launch_bounds__(256)
void conv1_kernel(const float* __restrict__ x, const float* __restrict__ w1,
                  const float* __restrict__ b1, _Float16* __restrict__ y1h)
{
    int idx = blockIdx.x * blockDim.x + threadIdx.x;
    if (idx >= BATCH * 16 * 44 * 44) return;
    int ox = idx % 44;
    int t  = idx / 44;
    int oy = t % 44;  t /= 44;
    int c  = t % 16;
    int b  = t / 16;
    const float* xb = x + (size_t)b * 92 * 92;
    const float* wc = w1 + c * 36;
    float s = b1[c];
#pragma unroll
    for (int ky = 0; ky < 6; ++ky)
#pragma unroll
        for (int kx = 0; kx < 6; ++kx)
            s += xb[(oy * 2 + ky) * 92 + ox * 2 + kx] * wc[ky * 6 + kx];
    s = (s > 0.f) ? s : 0.01f * s;
    y1h[idx] = (_Float16)s;
}

// ---------------------------------------------------------------- label statistics
__global__ __launch_bounds__(128)
void label_kernel(const float* __restrict__ x, float* __restrict__ feat,
                  _Float16* __restrict__ featH)
{
    __shared__ float rowsum[92];
    __shared__ float colsum[92];
    const int b = blockIdx.x;
    const float* xb = x + (size_t)b * 92 * 92;
    const int t = threadIdx.x;
    if (t < 92) {
        float rs = 0.f, cs = 0.f;
        for (int k = 0; k < 92; ++k) {
            rs += xb[t * 92 + k];   // xv.sum(2)
            cs += xb[k * 92 + t];   // xv.sum(1)
        }
        rowsum[t] = rs;
        colsum[t] = cs;
    }
    if (t >= 96 && t < 123) featH[(size_t)b * FEATPAD + 261 + (t - 96)] = (_Float16)0.f;
    __syncthreads();
    if (t == 0) {
        float sum = 0.f, sx = 0.f, sz = 0.f;
        for (int k = 0; k < 90; ++k) {
            float w = 0.5f + (float)k * (3.0f / 90.0f);
            sum += rowsum[1 + k];
            sx  += rowsum[1 + k] * w;
            sz  += colsum[1 + k] * w;
        }
        float xm = sx / sum, zm = sz / sum;
        float vx = 0.f, vz = 0.f;
        for (int k = 0; k < 90; ++k) {
            float w = 0.5f + (float)k * (3.0f / 90.0f);
            float a = rowsum[1 + k] * w - xm;
            float c = colsum[1 + k] * w - zm;
            vx += a * a;
            vz += c * c;
        }
        float lab[5] = { sum, xm, zm, sqrtf(vx) / sum, sqrtf(vz) / sum };
#pragma unroll
        for (int j = 0; j < 5; ++j) {
            feat [(size_t)b * FEAT    + 256 + j] = lab[j];
            featH[(size_t)b * FEATPAD + 256 + j] = (_Float16)lab[j];
        }
    }
}

// ---------------------------------------------------------------- implicit-GEMM conv via WMMA
// Block = NT waves, one 16-pixel M-tile. im2col A-tile staged once in LDS (shared by
// all NT n-tile waves); fragments = 2x ds_load_b128 (A) + 2x global b128 (padded weights).
template<int CIN, int KH, int KW, int H, int W, int OH, int OW, int COUT, int STRIDE, int MODE>
__global__ __launch_bounds__((COUT / 16) * 32)
void conv_wmma_kernel(const _Float16* __restrict__ inh, const _Float16* __restrict__ wgtP,
                      const float* __restrict__ bias, _Float16* __restrict__ outh,
                      float* __restrict__ feat, _Float16* __restrict__ featH)
{
    constexpr int KTOT = CIN * KH * KW;
    constexpr int KPAD = (KTOT + 31) / 32 * 32;
    constexpr int NT   = COUT / 16;

    __shared__ __align__(16) _Float16 Atile[16 * KPAD];

    const int tileM = blockIdx.x;
    const int tid   = threadIdx.x;

    // im2col gather (scattered u16 reads, done once per block)
    for (int idx = tid; idx < 16 * KPAD; idx += NT * 32) {
        const int pix = idx / KPAD;
        const int k   = idx % KPAD;
        _Float16 v = (_Float16)0.f;
        if (k < KTOT) {
            const int p    = tileM * 16 + pix;
            const int b    = p / (OH * OW);
            const int prem = p % (OH * OW);
            const int oy = prem / OW, ox = prem % OW;
            const int ci = k / (KH * KW);
            const int r  = k % (KH * KW);
            const int ky = r / KW, kx = r % KW;
            v = inh[(((size_t)b * CIN + ci) * H + oy * STRIDE + ky) * W + ox * STRIDE + kx];
        }
        Atile[idx] = v;
    }
    __syncthreads();

    const int lane = tid & 31;
    const int g = lane >> 4, l15 = lane & 15;
    const int w = tid >> 5;                 // n-tile index
    const int n = w * 16 + l15;
    const _Float16* wrow = wgtP + (size_t)n * KPAD;

    v8f acc = {0.f, 0.f, 0.f, 0.f, 0.f, 0.f, 0.f, 0.f};
    for (int kc = 0; kc < KPAD; kc += 32) {
        v16h av = make_frag(&Atile[l15 * KPAD + kc + 8 * g],
                            &Atile[l15 * KPAD + kc + 16 + 8 * g]);
        v16h bv = make_frag(wrow + kc + 8 * g, wrow + kc + 16 + 8 * g);
        acc = __builtin_amdgcn_wmma_f32_16x16x32_f16(false, av, false, bv,
                                                     (short)0, acc, false, false);
    }

    const float bn = bias[n];
#pragma unroll
    for (int r = 0; r < 8; ++r) {
        const int p2 = tileM * 16 + r + 8 * g;
        float v = acc[r] + bn;
        v = (v > 0.f) ? v : 0.01f * v;
        const int b2   = p2 / (OH * OW);
        const int rem2 = p2 % (OH * OW);
        if (MODE == 0) {
            const int oy2 = rem2 / OW, ox2 = rem2 % OW;
            outh[(((size_t)b2 * COUT + n) * OH + oy2) * OW + ox2] = (_Float16)v;
        } else {
            const int col = n * (OH * OW) + rem2;
            feat [(size_t)b2 * FEAT    + col] = v;
            featH[(size_t)b2 * FEATPAD + col] = (_Float16)v;
        }
    }
}

// ---------------------------------------------------------------- M = feature @ T  (WMMA + TDM)
// Block = 8 waves: one 16-row M-tile x 128 columns. B panel (128 rows of N-major ThT)
// staged into LDS by the Tensor Data Mover (D# per ISA 8.3/8.4); A tile copied with b128.
__global__ __launch_bounds__(256)
void gemm_MT_kernel(const _Float16* __restrict__ featH, const _Float16* __restrict__ ThT,
                    float* __restrict__ Mbuf)
{
    __shared__ __align__(16) _Float16 Bpanel[128 * FEATPAD];   // 72 KB
    __shared__ __align__(16) _Float16 Atile [16  * FEATPAD];   // 9 KB

    const int tileM = blockIdx.x >> 3;      // 32 M-tiles
    const int cg    = blockIdx.x & 7;       // 8 column groups of 128
    const int tid   = threadIdx.x;

#if USE_TDM
    if (tid < 32) {   // wave 0 issues one TDM 2D tile load: 288 x 128 halves
        const unsigned long long ga =
            (unsigned long long)(uintptr_t)(ThT + (size_t)cg * 128 * FEATPAD);
        const unsigned int lds = (unsigned int)(uintptr_t)Bpanel;
        v4u g0;
        g0[0] = 1u;                                            // count=1, no gather
        g0[1] = lds;                                           // lds_addr
        g0[2] = (unsigned int)ga;                              // global_addr[31:0]
        g0[3] = (unsigned int)((ga >> 32) & 0x1FFFFFFu) | (2u << 30);  // addr[56:32] | type=2
        v8i g1;
        g1[0] = (int)(1u << 16);            // workgroup_mask=0, data_size=1 (2 bytes)
        g1[1] = (int)(288u << 16);          // tensor_dim0[15:0]=288 (bits 79:64 are 0)
        g1[2] = (int)(128u << 16);          // tensor_dim1[15:0]=128
        g1[3] = (int)(288u << 16);          // tile_dim0=288
        g1[4] = (int)128;                   // tile_dim1=128, tile_dim2=0
        g1[5] = (int)288;                   // tensor_dim0_stride low 32 = 288
        g1[6] = 0;                          // stride hi / tensor_dim1_stride
        g1[7] = 0;
        v4i z4 = {0, 0, 0, 0};
        v8i z8 = {0, 0, 0, 0, 0, 0, 0, 0};
        // 6-arg variant (clang-23 / therock-10.0 headers): extra int32x8 group, zeroed.
        __builtin_amdgcn_tensor_load_to_lds(g0, g1, z4, z4, z8, 0);
        __builtin_amdgcn_s_wait_tensorcnt(0);
    }
#else
    {
        const v8h* src = (const v8h*)(ThT + (size_t)cg * 128 * FEATPAD);
        v8h* dst = (v8h*)Bpanel;
        for (int idx = tid; idx < 128 * FEATPAD / 8; idx += 256) dst[idx] = src[idx];
    }
#endif
    {
        const v8h* src = (const v8h*)(featH + (size_t)tileM * 16 * FEATPAD);
        v8h* dst = (v8h*)Atile;
        for (int idx = tid; idx < 16 * FEATPAD / 8; idx += 256) dst[idx] = src[idx];
    }
    __syncthreads();

    const int lane = tid & 31;
    const int g = lane >> 4, l15 = lane & 15;
    const int w = tid >> 5;

    v8f acc = {0.f, 0.f, 0.f, 0.f, 0.f, 0.f, 0.f, 0.f};
    for (int kc = 0; kc < FEATPAD; kc += 32) {
        v16h av = make_frag(&Atile[l15 * FEATPAD + kc + 8 * g],
                            &Atile[l15 * FEATPAD + kc + 16 + 8 * g]);
        v16h bv = make_frag(&Bpanel[(w * 16 + l15) * FEATPAD + kc + 8 * g],
                            &Bpanel[(w * 16 + l15) * FEATPAD + kc + 16 + 8 * g]);
        acc = __builtin_amdgcn_wmma_f32_16x16x32_f16(false, av, false, bv,
                                                     (short)0, acc, false, false);
    }

    const int n = cg * 128 + w * 16 + l15;
#pragma unroll
    for (int r = 0; r < 8; ++r)
        Mbuf[(size_t)(tileM * 16 + r + 8 * g) * MDIM + n] = acc[r];
}

// ---------------------------------------------------------------- pairwise L1 + exp-sum
__global__ __launch_bounds__(256)
void dist_kernel(const float* __restrict__ Mbuf, float* __restrict__ feat)
{
    __shared__ __align__(16) float Mi[MDIM];
    __shared__ float red[256];
    const int i = blockIdx.x;
    {
        const float4* src = (const float4*)(Mbuf + (size_t)i * MDIM);
        float4* dst = (float4*)Mi;
        for (int t = threadIdx.x; t < MDIM / 4; t += 256) dst[t] = src[t];
    }
    __syncthreads();
    const int c  = threadIdx.x & 63;
    const int jg = threadIdx.x >> 6;
    const float4* mi4 = (const float4*)&Mi[c * 16];
    const float4 a0 = mi4[0], a1 = mi4[1], a2 = mi4[2], a3 = mi4[3];
    float accv = 0.f;
    for (int j = jg; j < BATCH; j += 4) {
        const float4* mj = (const float4*)(Mbuf + (size_t)j * MDIM + c * 16);
        __builtin_prefetch(Mbuf + (size_t)((j + 4) & (BATCH - 1)) * MDIM + c * 16, 0, 1);
        const float4 b0 = mj[0], b1 = mj[1], b2 = mj[2], b3 = mj[3];
        float d = fabsf(a0.x - b0.x) + fabsf(a0.y - b0.y) + fabsf(a0.z - b0.z) + fabsf(a0.w - b0.w)
                + fabsf(a1.x - b1.x) + fabsf(a1.y - b1.y) + fabsf(a1.z - b1.z) + fabsf(a1.w - b1.w)
                + fabsf(a2.x - b2.x) + fabsf(a2.y - b2.y) + fabsf(a2.z - b2.z) + fabsf(a2.w - b2.w)
                + fabsf(a3.x - b3.x) + fabsf(a3.y - b3.y) + fabsf(a3.z - b3.z) + fabsf(a3.w - b3.w);
        accv += expf(d);
    }
    red[threadIdx.x] = accv;
    __syncthreads();
    if (jg == 0) {
        const float tot = red[c] + red[c + 64] + red[c + 128] + red[c + 192];
        feat[(size_t)i * FEAT + FEATK + c] = tot;   // cols 261..324
    }
}

// ---------------------------------------------------------------- sigmoid(z @ Wm + bm)
__global__ __launch_bounds__(256)
void final_kernel(const float* __restrict__ feat, const float* __restrict__ Wm,
                  const float* __restrict__ bm, float* __restrict__ out)
{
    const int row  = blockIdx.x * 8 + (threadIdx.x >> 5);
    const int lane = threadIdx.x & 31;
    if (row >= BATCH) return;
    float s = 0.f;
    for (int k = lane; k < FEAT; k += 32) s += feat[(size_t)row * FEAT + k] * Wm[k];
#pragma unroll
    for (int off = 16; off; off >>= 1) s += __shfl_down(s, off, 32);
    if (lane == 0) out[row] = 1.f / (1.f + expf(-(s + bm[0])));
}

// ---------------------------------------------------------------- launcher
extern "C" void kernel_launch(void* const* d_in, const int* in_sizes, int n_in,
                              void* d_out, int out_size, void* d_ws, size_t ws_size,
                              hipStream_t stream)
{
    const float* x  = (const float*)d_in[0];
    const float* w1 = (const float*)d_in[1];
    const float* b1 = (const float*)d_in[2];
    const float* w2 = (const float*)d_in[3];
    const float* b2 = (const float*)d_in[4];
    const float* w3 = (const float*)d_in[5];
    const float* b3 = (const float*)d_in[6];
    const float* w4 = (const float*)d_in[7];
    const float* b4 = (const float*)d_in[8];
    const float* T  = (const float*)d_in[9];
    const float* Wm = (const float*)d_in[10];
    const float* bm = (const float*)d_in[11];
    float* out = (float*)d_out;

    char* ws = (char*)d_ws;
    size_t off = 0;
    auto carve = [&](size_t bytes) -> char* {
        char* p = ws + off;
        off += (bytes + 255) & ~(size_t)255;
        return p;
    };
    _Float16* y1h   = (_Float16*)carve(sizeof(_Float16) * (size_t)BATCH * 16 * 44 * 44);
    _Float16* y2h   = (_Float16*)carve(sizeof(_Float16) * (size_t)BATCH * 32 * 14 * 14);
    _Float16* y3h   = (_Float16*)carve(sizeof(_Float16) * (size_t)BATCH * 48 * 4 * 4);
    _Float16* w2p   = (_Float16*)carve(sizeof(_Float16) * 32 * 416);
    _Float16* w3p   = (_Float16*)carve(sizeof(_Float16) * 48 * 800);
    _Float16* w4p   = (_Float16*)carve(sizeof(_Float16) * 64 * 448);
    _Float16* ThT   = (_Float16*)carve(sizeof(_Float16) * MDIM * FEATPAD);
    _Float16* featH = (_Float16*)carve(sizeof(_Float16) * (size_t)BATCH * FEATPAD);
    float*    feat  = (float*)   carve(sizeof(float)    * (size_t)BATCH * FEAT);
    float*    Mbuf  = (float*)   carve(sizeof(float)    * (size_t)BATCH * MDIM);

    prep_kernel<<<(MDIM * FEATPAD + 255) / 256, 256, 0, stream>>>(w2, w3, w4, T,
                                                                  w2p, w3p, w4p, ThT);
    conv1_kernel<<<(BATCH * 16 * 44 * 44 + 255) / 256, 256, 0, stream>>>(x, w1, b1, y1h);
    label_kernel<<<BATCH, 128, 0, stream>>>(x, feat, featH);

    // conv2: 16->32, 5x5 s3, 44->14 ; 6272 M-tiles, 2 waves/block
    conv_wmma_kernel<16, 5, 5, 44, 44, 14, 14, 32, 3, 0>
        <<<6272, 64, 0, stream>>>(y1h, w2p, b2, y2h, nullptr, nullptr);
    // conv3: 32->48, 5x5 s3, 14->4 ; 512 M-tiles, 3 waves/block
    conv_wmma_kernel<32, 5, 5, 14, 14, 4, 4, 48, 3, 0>
        <<<512, 96, 0, stream>>>(y2h, w3p, b3, y3h, nullptr, nullptr);
    // conv4: 48->64, 3x3 s1, 4->2 ; 128 M-tiles, 4 waves/block ; writes feature
    conv_wmma_kernel<48, 3, 3, 4, 4, 2, 2, 64, 1, 1>
        <<<128, 128, 0, stream>>>(y3h, w4p, b4, nullptr, feat, featH);

    gemm_MT_kernel<<<256, 256, 0, stream>>>(featH, ThT, Mbuf);
    dist_kernel<<<BATCH, 256, 0, stream>>>(Mbuf, feat);
    final_kernel<<<BATCH / 8, 256, 0, stream>>>(feat, Wm, bm, out);
}
